// GlobalAttn_20349555048606
// MI455X (gfx1250) — matrix operand order repaired and checked by reading, not compiled, not run on hardware
//
#include <hip/hip_runtime.h>
#include <hip/hip_bf16.h>

#define NN 32768
#define DD 1024
#define HH 8
#define HCC 128
#define LL 3
#define KVN 144           // 128 kv cols + den col + pad to 16
#define SPLITS 32
#define CHUNK (NN / SPLITS)
#define LN_EPS 1e-5f

typedef __attribute__((ext_vector_type(16))) __bf16 v16bf;
typedef __attribute__((ext_vector_type(8)))  float  v8f;
typedef unsigned int u32x4 __attribute__((ext_vector_type(4)));
typedef int          i32x4 __attribute__((ext_vector_type(4)));
typedef int          i32x8 __attribute__((ext_vector_type(8)));

#if defined(__gfx1250__) && __has_builtin(__builtin_amdgcn_tensor_load_to_lds)
#define HAVE_TDM 1
#else
#define HAVE_TDM 0
#endif

__device__ __forceinline__ unsigned short f2bf(float f) {
  union { float f; unsigned u; } x; x.f = f;
  unsigned r = x.u + 0x7FFFu + ((x.u >> 16) & 1u);   // round-to-nearest-even
  return (unsigned short)(r >> 16);
}
__device__ __forceinline__ float bf2f(unsigned short s) {
  union { unsigned u; float f; } x; x.u = ((unsigned)s) << 16;
  return x.f;
}

union FragU { unsigned u[8]; v16bf v; };

// A-matrix 16x32 bf16 fragment (ISA 7.12.2): lane row = l%16, K-half by lane>=16.
__device__ __forceinline__ v16bf load_a_frag(const unsigned short* base, int ld,
                                             int row, int kb) {
  FragU f;
  const unsigned short* p = base + row * ld;
#pragma unroll
  for (int j = 0; j < 4; ++j) f.u[j]     = *(const unsigned*)(p + kb + 2 * j);
#pragma unroll
  for (int j = 0; j < 4; ++j) f.u[4 + j] = *(const unsigned*)(p + 16 + kb + 2 * j);
  return f.v;
}
// B-matrix 32x16 bf16 fragment: lane col = l%16, K = hi*8 + 2j (+16 for j>=4). LDS holds B^T [n][k].
__device__ __forceinline__ v16bf load_b_frag(const unsigned short* base, int ld,
                                             int col, int kb) {
  FragU f;
  const unsigned short* p = base + col * ld;
#pragma unroll
  for (int j = 0; j < 4; ++j) f.u[j]     = *(const unsigned*)(p + kb + 2 * j);
#pragma unroll
  for (int j = 0; j < 4; ++j) f.u[4 + j] = *(const unsigned*)(p + 16 + kb + 2 * j);
  return f.v;
}

__device__ __forceinline__ void wait_tensorcnt0() {
#if __has_builtin(__builtin_amdgcn_s_wait_tensorcnt)
  __builtin_amdgcn_s_wait_tensorcnt(0);
#else
  asm volatile("s_wait_tensorcnt 0x0" ::: "memory");
#endif
}

#if HAVE_TDM
// 2D bf16 tile -> LDS via Tensor Data Mover (ISA ch.8).  Rows of tile0 elems,
// tile1 rows, row stride = stride0 elems.  LDS pad: 1 DWORD per 16 DWORDs
// (64B row -> 68B = 34-ushort padded rows, matching BKP).
__device__ __forceinline__ void tdm_load_2d(unsigned lds_off, const void* gptr,
                                            unsigned tile0, unsigned tile1,
                                            unsigned stride0) {
  unsigned long long ga = (unsigned long long)(uintptr_t)gptr;
  unsigned lds  = (unsigned)__builtin_amdgcn_readfirstlane((int)lds_off);
  unsigned galo = (unsigned)__builtin_amdgcn_readfirstlane((int)(unsigned)ga);
  unsigned gahi = (unsigned)__builtin_amdgcn_readfirstlane((int)(unsigned)(ga >> 32));
  unsigned str  = (unsigned)__builtin_amdgcn_readfirstlane((int)stride0);

  u32x4 g0;
  g0.x = 1u;                               // count=1, user mode
  g0.y = lds;                              // lds_addr (bytes)
  g0.z = galo;                             // global_addr[31:0]
  g0.w = (gahi & 0x01FFFFFFu) | (2u << 30);// global_addr[56:32] | type=2

  const unsigned td0 = 0x40000000u, td1 = 0x40000000u;  // huge dims: no OOB clip
  i32x8 g1;
  g1[0] = (int)((1u << 16) | (1u << 20) | (3u << 22));  // data_size=2B, pad 1dw/16dw
  g1[1] = (int)((td0 & 0xFFFFu) << 16);                 // tensor_dim0[15:0]
  g1[2] = (int)((td0 >> 16) | ((td1 & 0xFFFFu) << 16)); // dim0 hi | dim1 lo
  g1[3] = (int)((td1 >> 16) | (tile0 << 16));           // dim1 hi | tile_dim0
  g1[4] = (int)(tile1 & 0xFFFFu);                       // tile_dim1 (tile_dim2=0)
  g1[5] = (int)str;                                     // tensor_dim0_stride lo
  g1[6] = 0;                                            // stride hi | dim1_stride lo
  g1[7] = 0;
  i32x4 gz = {0, 0, 0, 0};
#if __clang_major__ >= 23
  i32x8 gz8 = {0, 0, 0, 0, 0, 0, 0, 0};
  __builtin_amdgcn_tensor_load_to_lds(g0, g1, gz, gz, gz8, 0);
#else
  __builtin_amdgcn_tensor_load_to_lds(g0, g1, gz, gz, 0);
#endif
}
#endif

// ---------------------------------------------------------------------------
// C = epilogue(A[MxK] @ Bt[NxK]^T + bias).  All bf16 in global, f32 accum.
// EPI: 0=none, 1=sigmoid, 2=relu.  OUT16: bf16 (1) or f32 (0) output.
// Block 128x128, BK=32; 8 waves (4x2), wave tile 32x64 = 2x4 WMMA tiles.
// ---------------------------------------------------------------------------
template <int EPI, int OUT16>
__global__ __launch_bounds__(256)
void gemm_bf16(const unsigned short* __restrict__ A,
               const unsigned short* __restrict__ Bt,
               const float* __restrict__ bias,
               unsigned short* __restrict__ C16, float* __restrict__ C32,
               int M, int Nc, int K) {
  constexpr int BM = 128, BN = 128, BK = 32, BKP = 34;
  __shared__ unsigned short As[BM][BKP];
  __shared__ unsigned short Bs[BN][BKP];
  const int tid = threadIdx.x;
  const int m0 = blockIdx.y * BM, n0 = blockIdx.x * BN;
  const int lane = tid & 31, wid = tid >> 5;
  const int l16 = lane & 15, hi = lane >> 4;
  const int wm = (wid & 3) * 32, wn = (wid >> 2) * 64;
  v8f acc[2][4] = {};

  for (int k0 = 0; k0 < K; k0 += BK) {
    __syncthreads();   // previous tile fully consumed before overwrite
#if HAVE_TDM
    if (wid == 0) {
      tdm_load_2d((unsigned)(uintptr_t)&As[0][0],
                  A + (size_t)m0 * K + k0, BK, BM, (unsigned)K);
      tdm_load_2d((unsigned)(uintptr_t)&Bs[0][0],
                  Bt + (size_t)n0 * K + k0, BK, BN, (unsigned)K);
      wait_tensorcnt0();
    }
#else
#pragma unroll
    for (int it = 0; it < 8; ++it) {
      int slot = tid + it * 256;            // 2048 dword slots = 128x32 ushorts
      int row = slot >> 4;
      int kc = (slot & 15) * 2;
      *(unsigned*)&As[row][kc] =
          *(const unsigned*)(A + (size_t)(m0 + row) * K + k0 + kc);
      *(unsigned*)&Bs[row][kc] =
          *(const unsigned*)(Bt + (size_t)(n0 + row) * K + k0 + kc);
    }
#endif
    __syncthreads();
    v16bf af[2], bfr[4];
#pragma unroll
    for (int r = 0; r < 2; ++r)
      af[r] = load_a_frag(&As[0][0], BKP, wm + r * 16 + l16, hi * 8);
#pragma unroll
    for (int c = 0; c < 4; ++c)
      bfr[c] = load_b_frag(&Bs[0][0], BKP, wn + c * 16 + l16, hi * 8);
#pragma unroll
    for (int r = 0; r < 2; ++r)
#pragma unroll
      for (int c = 0; c < 4; ++c)
        acc[r][c] = __builtin_amdgcn_wmma_f32_16x16x32_bf16(
            false, af[r], false, bfr[c], (short)0, acc[r][c], false, false);
  }

#pragma unroll
  for (int r = 0; r < 2; ++r)
#pragma unroll
    for (int c = 0; c < 4; ++c) {
      int col = n0 + wn + c * 16 + l16;
      float bv = bias[col];
#pragma unroll
      for (int p = 0; p < 8; ++p) {
        int row = m0 + wm + r * 16 + p + hi * 8;
        float v = acc[r][c][p] + bv;
        if (EPI == 1) v = 1.f / (1.f + __expf(-v));
        if (EPI == 2) v = fmaxf(v, 0.f);
        if (OUT16) C16[(size_t)row * Nc + col] = f2bf(v);
        else       C32[(size_t)row * Nc + col] = v;
      }
    }
}

// ---------------------------------------------------------------------------
// Split-K partials of kv[h] = K_h^T @ [V_h | 1 | 0...] (col 128 => k_sum).
// ---------------------------------------------------------------------------
__global__ __launch_bounds__(256)
void kv_partial(const unsigned short* __restrict__ Km,
                const unsigned short* __restrict__ Vm,
                float* __restrict__ KVp) {
  constexpr int BK = 32, BKP = 34;
  __shared__ unsigned short As[HCC][BKP];  // A[d][n] = K[nb+n][d*8+h]
  __shared__ unsigned short Bs[KVN][BKP];  // B^T[m][n]
  const int tid = threadIdx.x;
  const int h = blockIdx.x & (HH - 1);
  const int s = blockIdx.x / HH;
  const int nb0 = s * CHUNK;
  const int lane = tid & 31, wid = tid >> 5;
  const int l16 = lane & 15, hi = lane >> 4;
  const int m0 = wid * 16;
  v8f acc[9] = {};

  for (int n0 = 0; n0 < CHUNK; n0 += BK) {
    const int nb = nb0 + n0;
#pragma unroll
    for (int it = 0; it < 16; ++it) {
      int idx = tid + it * 256;
      int d = idx & 127, n = idx >> 7;
      As[d][n] = Km[(size_t)(nb + n) * DD + d * HH + h];
    }
#pragma unroll
    for (int it = 0; it < 18; ++it) {
      int idx = tid + it * 256;
      int m = idx % KVN, n = idx / KVN;
      Bs[m][n] = (m < HCC) ? Vm[(size_t)(nb + n) * DD + m * HH + h]
                           : (m == HCC ? (unsigned short)0x3F80u
                                       : (unsigned short)0u);
    }
    __syncthreads();
    v16bf af = load_a_frag(&As[0][0], BKP, m0 + l16, hi * 8);
#pragma unroll
    for (int c = 0; c < 9; ++c) {
      v16bf bfr = load_b_frag(&Bs[0][0], BKP, c * 16 + l16, hi * 8);
      acc[c] = __builtin_amdgcn_wmma_f32_16x16x32_bf16(
          false, af, false, bfr, (short)0, acc[c], false, false);
    }
    __syncthreads();
  }

  float* out = KVp + (size_t)blockIdx.x * HCC * KVN;
#pragma unroll
  for (int c = 0; c < 9; ++c)
#pragma unroll
    for (int p = 0; p < 8; ++p)
      out[(m0 + p + hi * 8) * KVN + c * 16 + l16] = acc[c][p];
}

__global__ __launch_bounds__(256)
void kv_reduce(const float* __restrict__ KVp, float* __restrict__ KV) {
  const int per = HCC * KVN;
  int idx = blockIdx.x * 256 + threadIdx.x;
  if (idx >= HH * per) return;
  int h = idx / per, inner = idx % per;
  float a = 0.f;
#pragma unroll 4
  for (int s = 0; s < SPLITS; ++s) a += KVp[(size_t)(s * HH + h) * per + inner];
  KV[idx] = a;
}

// ---------------------------------------------------------------------------
// num = q @ kv[h]; col 128 = den = q.k_sum; XI = num/den (bf16, aliases Km:
// disjoint rows x (col mod 8) across blocks, reads precede writes in-block).
// ---------------------------------------------------------------------------
__global__ __launch_bounds__(256)
void num_div(const unsigned short* __restrict__ Km, const float* __restrict__ KV,
             unsigned short* __restrict__ XI) {
  constexpr int BK = 32, BKP = 34;
  __shared__ unsigned short Qs[128][BKP];
  __shared__ unsigned short Bs[KVN][BKP];
  __shared__ float denS[128];
  const int tid = threadIdx.x;
  const int h = blockIdx.x;
  const int n0 = blockIdx.y * 128;
  const int lane = tid & 31, wid = tid >> 5;
  const int l16 = lane & 15, hi = lane >> 4;
  const int m0 = wid * 16;
  v8f acc[9] = {};

  for (int k0 = 0; k0 < HCC; k0 += BK) {
#pragma unroll
    for (int it = 0; it < 16; ++it) {
      int idx = tid + it * 256;
      int r = idx & 127, kk = idx >> 7;
      Qs[r][kk] = Km[(size_t)(n0 + r) * DD + (k0 + kk) * HH + h];
    }
#pragma unroll
    for (int it = 0; it < 18; ++it) {
      int idx = tid + it * 256;
      int m = idx % KVN, kk = idx / KVN;
      Bs[m][kk] = f2bf(KV[(size_t)(h * HCC + k0 + kk) * KVN + m]);
    }
    __syncthreads();
    v16bf af = load_a_frag(&Qs[0][0], BKP, m0 + l16, hi * 8);
#pragma unroll
    for (int c = 0; c < 9; ++c) {
      v16bf bfr = load_b_frag(&Bs[0][0], BKP, c * 16 + l16, hi * 8);
      acc[c] = __builtin_amdgcn_wmma_f32_16x16x32_bf16(
          false, af, false, bfr, (short)0, acc[c], false, false);
    }
    __syncthreads();
  }

  if (l16 == 0) {
#pragma unroll
    for (int p = 0; p < 8; ++p) denS[m0 + p + hi * 8] = acc[8][p];
  }
  __syncthreads();

#pragma unroll
  for (int c = 0; c < 8; ++c)
#pragma unroll
    for (int p = 0; p < 8; ++p) {
      int row = m0 + p + hi * 8;
      XI[(size_t)(n0 + row) * DD + (c * 16 + l16) * HH + h] =
          f2bf(acc[c][p] / denS[row]);
    }
}

// ---------------------------------------------------------------------------
// In-place LayerNorm over D fused with *(h + beta).  bf16 in/out.
// ---------------------------------------------------------------------------
__global__ __launch_bounds__(256)
void ln_mul(unsigned short* __restrict__ XI, const unsigned short* __restrict__ Hm,
            const float* __restrict__ g, const float* __restrict__ b,
            const float* __restrict__ beta) {
  __shared__ float rs[256], rq[256];
  const int tid = threadIdx.x;
  const size_t row = blockIdx.x;
  float x[4]; float s = 0.f, q = 0.f;
#pragma unroll
  for (int j = 0; j < 4; ++j) {
    x[j] = bf2f(XI[row * DD + tid + j * 256]);
    s += x[j]; q += x[j] * x[j];
  }
  rs[tid] = s; rq[tid] = q; __syncthreads();
  for (int off = 128; off > 0; off >>= 1) {
    if (tid < off) { rs[tid] += rs[tid + off]; rq[tid] += rq[tid + off]; }
    __syncthreads();
  }
  const float mu = rs[0] * (1.f / DD);
  const float var = rq[0] * (1.f / DD) - mu * mu;
  const float rstd = rsqrtf(var + LN_EPS);
#pragma unroll
  for (int j = 0; j < 4; ++j) {
    int c = tid + j * 256;
    float y = g[c] * (x[j] - mu) * rstd + b[c];
    XI[row * DD + c] = f2bf(y * (bf2f(Hm[row * DD + c]) + beta[c]));
  }
}

// fp32 -> bf16 elementwise (4 per thread)
__global__ __launch_bounds__(256)
void cvt_bf16(const float* __restrict__ in, unsigned short* __restrict__ out) {
  size_t i = ((size_t)blockIdx.x * 256 + threadIdx.x) * 4;
  const float4 f = *(const float4*)(in + i);
  out[i + 0] = f2bf(f.x); out[i + 1] = f2bf(f.y);
  out[i + 2] = f2bf(f.z); out[i + 3] = f2bf(f.w);
}

// W[k][n] fp32 -> Wt[n][k] bf16 (32x32 LDS tile transpose)
__global__ __launch_bounds__(256)
void cvt_wt(const float* __restrict__ W, unsigned short* __restrict__ Wt) {
  __shared__ float tile[32][33];
  const int tx = threadIdx.x & 31, ty = threadIdx.x >> 5;
  const int bx = blockIdx.x * 32, by = blockIdx.y * 32;
#pragma unroll
  for (int j = 0; j < 32; j += 8)
    tile[ty + j][tx] = W[(size_t)(by + ty + j) * DD + bx + tx];
  __syncthreads();
#pragma unroll
  for (int j = 0; j < 32; j += 8)
    Wt[(size_t)(bx + ty + j) * DD + by + tx] = f2bf(tile[tx][ty + j]);
}

// ---------------------------------------------------------------------------
extern "C" void kernel_launch(void* const* d_in, const int* in_sizes, int n_in,
                              void* d_out, int out_size, void* d_ws, size_t ws_size,
                              hipStream_t stream) {
  const float* x  = (const float*)d_in[0];
  const float* Wh = (const float*)d_in[1];
  const float* bh = (const float*)d_in[2];
  const float* Wk = (const float*)d_in[3];
  const float* bk = (const float*)d_in[4];
  const float* Wv = (const float*)d_in[5];
  const float* bv = (const float*)d_in[6];
  const float* lg = (const float*)d_in[7];
  const float* lb = (const float*)d_in[8];
  const float* be = (const float*)d_in[9];
  const float* Wo = (const float*)d_in[10];
  const float* bo = (const float*)d_in[11];
  float* out = (float*)d_out;

  const size_t ND = (size_t)NN * DD, WSZ = (size_t)DD * DD;
  unsigned short* X0  = (unsigned short*)d_ws;  // bf16 activations
  unsigned short* X1  = X0 + ND;
  unsigned short* X2  = X1 + ND;
  unsigned short* Hm  = X2 + ND;
  unsigned short* Km  = Hm + ND;                // k; later xi (aliased)
  unsigned short* Vm  = Km + ND;
  unsigned short* WhT = Vm + ND;                // bf16 transposed weights
  unsigned short* WkT = WhT + WSZ;
  unsigned short* WvT = WkT + WSZ;
  unsigned short* WoT = WvT + WSZ;
  float* KVp = (float*)(WoT + WSZ);
  float* KV  = KVp + (size_t)SPLITS * HH * HCC * KVN;

  const dim3 blk(256);
  const dim3 gGemm(DD / 128, NN / 128);
  const dim3 gWt(DD / 32, DD / 32);

  cvt_bf16<<<dim3(ND / 1024), blk, 0, stream>>>(x, X0);
  cvt_wt<<<gWt, blk, 0, stream>>>(Wo, WoT);

  for (int i = 0; i < LL; ++i) {
    const unsigned short* xin = (i == 0) ? X0 : (i == 1 ? X1 : X2);
    unsigned short* xnext16 = (i == 0) ? X1 : X2;
    const size_t wofs = (size_t)i * WSZ, bofs = (size_t)i * DD;

    cvt_wt<<<gWt, blk, 0, stream>>>(Wh + wofs, WhT);
    cvt_wt<<<gWt, blk, 0, stream>>>(Wk + wofs, WkT);
    cvt_wt<<<gWt, blk, 0, stream>>>(Wv + wofs, WvT);

    gemm_bf16<0, 1><<<gGemm, blk, 0, stream>>>(xin, WhT, bh + bofs, Hm, nullptr, NN, DD, DD);
    gemm_bf16<1, 1><<<gGemm, blk, 0, stream>>>(xin, WkT, bk + bofs, Km, nullptr, NN, DD, DD);
    gemm_bf16<0, 1><<<gGemm, blk, 0, stream>>>(xin, WvT, bv + bofs, Vm, nullptr, NN, DD, DD);

    kv_partial<<<dim3(HH * SPLITS), blk, 0, stream>>>(Km, Vm, KVp);
    kv_reduce<<<dim3((HH * HCC * KVN + 255) / 256), blk, 0, stream>>>(KVp, KV);

    num_div<<<dim3(HH, NN / 128), blk, 0, stream>>>(Km, KV, Km);
    ln_mul<<<dim3(NN), blk, 0, stream>>>(Km, Hm, lg + bofs, lb + bofs, be + bofs);

    if (i < LL - 1)
      gemm_bf16<2, 1><<<gGemm, blk, 0, stream>>>(Km, WoT, bo, xnext16, nullptr, NN, DD, DD);
    else
      gemm_bf16<2, 0><<<gGemm, blk, 0, stream>>>(Km, WoT, bo, nullptr, out, NN, DD, DD);
  }
}
